// Model_29162827940501
// MI455X (gfx1250) — matrix operand order repaired and checked by reading, not compiled, not run on hardware
//
#include <hip/hip_runtime.h>
#include <hip/hip_bf16.h>
#include <stdint.h>

typedef __attribute__((ext_vector_type(16))) __bf16 v16bf;
typedef __attribute__((ext_vector_type(8)))  __bf16 v8bf;
typedef __attribute__((ext_vector_type(8)))  float  v8f;

#define NUM_STATES 4096
#define NROWS      65536      // 4096 states * 16 balls

// ---------------- fast branchless activations (native TRANS ops) ----------------

__device__ __forceinline__ float fast_tanh(float x) {
#if __has_builtin(__builtin_amdgcn_tanhf)
    return __builtin_amdgcn_tanhf(x);           // v_tanh_f32 (gfx1250 TRANS op)
#else
    float cx = fminf(fmaxf(x, -10.0f), 10.0f);
    float e  = __builtin_amdgcn_exp2f(cx * 2.8853900817779268f);   // e^(2x)
    return (e - 1.0f) * __builtin_amdgcn_rcpf(e + 1.0f);
#endif
}

__device__ __forceinline__ float fast_sigmoid(float x) {
    float e = __builtin_amdgcn_exp2f(-x * 1.4426950408889634f);
    return __builtin_amdgcn_rcpf(1.0f + e);
}

__device__ __forceinline__ float fast_exp(float x) {
    return __builtin_amdgcn_exp2f(x * 1.4426950408889634f);
}

// ---------------- pack kernels: f32 -> bf16, K padded ----------------

__global__ __launch_bounds__(256) void pack_balls(const float* __restrict__ balls,
                                                  __hip_bfloat16* __restrict__ Abf) {
    int t = blockIdx.x * blockDim.x + threadIdx.x;           // over NROWS*64
    int row = t >> 6, k = t & 63;
    float v = (k < 36) ? balls[row * 36 + k] : 0.0f;
    Abf[t] = __float2bfloat16(v);
}

// W [K x Nn] f32 (fan_in x fan_out) -> Wt [Nn x Kp] bf16 (transposed, K padded)
__global__ __launch_bounds__(256) void pack_wt(const float* __restrict__ W,
                                               __hip_bfloat16* __restrict__ Wt,
                                               int K, int Nn, int Kp) {
    int t = blockIdx.x * blockDim.x + threadIdx.x;           // over Nn*Kp
    if (t >= Nn * Kp) return;
    int n = t / Kp, k = t % Kp;
    float v = (k < K) ? W[k * Nn + n] : 0.0f;
    Wt[t] = __float2bfloat16(v);
}

// ---------------- fragment helpers (ISA 7.12.2 layouts) ----------------

// A 16x32 bf16 fragment: lane holds row m=lane&15; K chunks at 8*half and 16+8*half
__device__ __forceinline__ v16bf load_afrag(const __hip_bfloat16* __restrict__ p, int half) {
    v8bf lo = *(const v8bf*)(p + 8 * half);
    v8bf hi = *(const v8bf*)(p + 16 + 8 * half);
    v16bf r;
#pragma unroll
    for (int i = 0; i < 8; ++i) { r[i] = lo[i]; r[i + 8] = hi[i]; }
    return r;
}

// B 32x16 bf16 fragment: lane holds column n=lane&15; K chunks at 8*half and 16+8*half
__device__ __forceinline__ v16bf make_bfrag(v8bf lo, v8bf hi) {
    v16bf r;
#pragma unroll
    for (int i = 0; i < 8; ++i) { r[i] = lo[i]; r[i + 8] = hi[i]; }
    return r;
}

#define WMMA_BF16(a, b, c) \
    __builtin_amdgcn_wmma_f32_16x16x32_bf16(false, (a), false, (b), (short)0, (c), false, false)

#define WAIT_ASYNC() asm volatile("s_wait_asynccnt 0" ::: "memory")

// Issue async copy of a 64(N) x KS(K) B tile into LDS (ASYNCcnt path, no wait).
// STRIDE chosen so rows are 16B-aligned and bank-conflict-free.
template<int KS, int STRIDE>
__device__ __forceinline__ void stage_issue(const __hip_bfloat16* __restrict__ Wt,
                                            __hip_bfloat16* Bs,
                                            int n0, int ks, int Kp, int tid) {
    constexpr int CPR = KS / 8;          // 16B chunks per row (8 or 16)
    constexpr int NCH = 64 * CPR;        // total chunks (512 or 1024)
#pragma unroll
    for (int i = 0; i < NCH / 256; ++i) {
        int c   = tid + i * 256;
        int row = c / CPR;               // column of B == row of Wt
        int cc  = c % CPR;               // 16B chunk within row
        const __hip_bfloat16* g = Wt + (size_t)(n0 + row) * Kp + ks + cc * 8;
        unsigned lds = (unsigned)(uintptr_t)&Bs[row * STRIDE + cc * 8];
        asm volatile("global_load_async_to_lds_b128 %0, %1, off"
                     :: "v"(lds), "v"(g) : "memory");
    }
}

// One kk-step: 2 A frags (global) x 4 B frags (LDS) -> 8 WMMAs.
// All B frags loaded up-front so ds_load latency overlaps the WMMA stream.
template<int STRIDE>
__device__ __forceinline__ void mma_step(const __hip_bfloat16* __restrict__ arow0,
                                         const __hip_bfloat16* __restrict__ arow1,
                                         const __hip_bfloat16* Bcur,
                                         int koff, int kk, int mrow, int half,
                                         v8f acc[2][4]) {
    v16bf a0 = load_afrag(arow0 + koff, half);
    v16bf a1 = load_afrag(arow1 + koff, half);
    v16bf bfr[4];
#pragma unroll
    for (int nb = 0; nb < 4; ++nb) {
        const __hip_bfloat16* bp = &Bcur[(nb * 16 + mrow) * STRIDE + kk + 8 * half];
        bfr[nb] = make_bfrag(*(const v8bf*)bp, *(const v8bf*)(bp + 16));
    }
#pragma unroll
    for (int nb = 0; nb < 4; ++nb) {
        acc[0][nb] = WMMA_BF16(a0, bfr[nb], acc[0][nb]);
        acc[1][nb] = WMMA_BF16(a1, bfr[nb], acc[1][nb]);
    }
}

// ---------------- GEMM: out = tanh(A @ Wt^T + bias), bf16 out ----------------
// Block: 8 waves, 256 M-rows; wave tile = 32(M) x 64(N); B double-buffered in LDS.
template<int KS>
__global__ __launch_bounds__(256) void gemm_tanh_bf16(const __hip_bfloat16* __restrict__ A,
                                                      const __hip_bfloat16* __restrict__ Wt,
                                                      const float* __restrict__ bias,
                                                      __hip_bfloat16* __restrict__ out,
                                                      int Mtot, int Kp, int Nn) {
    constexpr int STRIDE = KS + 8;       // 72 or 136 elems: 16B-aligned, conflict-free
    constexpr int BUFE   = 64 * STRIDE;
    __shared__ __hip_bfloat16 Bs[2 * BUFE];
    const int tid  = threadIdx.x;
    const int lane = tid & 31;
    const int w    = tid >> 5;
    const int mrow = lane & 15;
    const int half = lane >> 4;
    const size_t mbase = (size_t)blockIdx.x * 256 + (size_t)w * 32;
    const __hip_bfloat16* arow0 = A + (mbase + mrow) * Kp;
    const __hip_bfloat16* arow1 = arow0 + (size_t)16 * Kp;

    for (int n0 = 0; n0 < Nn; n0 += 64) {
        v8f acc[2][4];
#pragma unroll
        for (int f = 0; f < 2; ++f)
#pragma unroll
            for (int nb = 0; nb < 4; ++nb) acc[f][nb] = (v8f){0,0,0,0,0,0,0,0};

        __syncthreads();                              // buf0 readers (prev n0) done
        stage_issue<KS, STRIDE>(Wt, Bs, n0, 0, Kp, tid);
        int buf = 0;
        for (int ks = 0; ks < Kp; ks += KS) {
            WAIT_ASYNC();                             // my chunks for `buf` landed
            __syncthreads();                          // everyone's chunks visible
            if (ks + KS < Kp)                         // prefetch next stage
                stage_issue<KS, STRIDE>(Wt, Bs + (buf ^ 1) * BUFE, n0, ks + KS, Kp, tid);
            const __hip_bfloat16* Bcur = Bs + buf * BUFE;
#pragma unroll
            for (int kk = 0; kk < KS; kk += 32)
                mma_step<STRIDE>(arow0, arow1, Bcur, ks + kk, kk, mrow, half, acc);
            buf ^= 1;
        }
        // D layout: VGPR r holds (M = f*16 + r + 8*half, N = n0 + nb*16 + mrow)
#pragma unroll
        for (int f = 0; f < 2; ++f)
#pragma unroll
            for (int nb = 0; nb < 4; ++nb) {
                int col = n0 + nb * 16 + mrow;
                float bv = bias[col];
#pragma unroll
                for (int r = 0; r < 8; ++r) {
                    size_t m = mbase + f * 16 + 8 * half + r;
                    out[m * Nn + col] = __float2bfloat16(fast_tanh(acc[f][nb][r] + bv));
                }
            }
    }
}

// ---------------- GEMM3 fused: tanh + per-state max-pool ----------------
// Wave tile = 32 rows = 2 states. v [NUM_STATES x Nn] f32.
template<int KS>
__global__ __launch_bounds__(256) void gemm_tanh_pool(const __hip_bfloat16* __restrict__ A,
                                                      const __hip_bfloat16* __restrict__ Wt,
                                                      const float* __restrict__ bias,
                                                      float* __restrict__ v,
                                                      int Mtot, int Kp, int Nn) {
    constexpr int STRIDE = KS + 8;
    constexpr int BUFE   = 64 * STRIDE;
    __shared__ __hip_bfloat16 Bs[2 * BUFE];
    const int tid  = threadIdx.x;
    const int lane = tid & 31;
    const int w    = tid >> 5;
    const int mrow = lane & 15;
    const int half = lane >> 4;
    const size_t mbase = (size_t)blockIdx.x * 256 + (size_t)w * 32;
    const int state0 = (int)(mbase >> 4);
    const __hip_bfloat16* arow0 = A + (mbase + mrow) * Kp;
    const __hip_bfloat16* arow1 = arow0 + (size_t)16 * Kp;

    for (int n0 = 0; n0 < Nn; n0 += 64) {
        v8f acc[2][4];
#pragma unroll
        for (int f = 0; f < 2; ++f)
#pragma unroll
            for (int nb = 0; nb < 4; ++nb) acc[f][nb] = (v8f){0,0,0,0,0,0,0,0};

        __syncthreads();
        stage_issue<KS, STRIDE>(Wt, Bs, n0, 0, Kp, tid);
        int buf = 0;
        for (int ks = 0; ks < Kp; ks += KS) {
            WAIT_ASYNC();
            __syncthreads();
            if (ks + KS < Kp)
                stage_issue<KS, STRIDE>(Wt, Bs + (buf ^ 1) * BUFE, n0, ks + KS, Kp, tid);
            const __hip_bfloat16* Bcur = Bs + buf * BUFE;
#pragma unroll
            for (int kk = 0; kk < KS; kk += 32)
                mma_step<STRIDE>(arow0, arow1, Bcur, ks + kk, kk, mrow, half, acc);
            buf ^= 1;
        }
        // tanh + max over the 16 rows of each sub-tile (f) = one state
#pragma unroll
        for (int f = 0; f < 2; ++f)
#pragma unroll
            for (int nb = 0; nb < 4; ++nb) {
                int col = n0 + nb * 16 + mrow;
                float bv = bias[col];
                float mx = -3.402823466e+38f;
#pragma unroll
                for (int r = 0; r < 8; ++r) mx = fmaxf(mx, fast_tanh(acc[f][nb][r] + bv));
                float other = __shfl_xor(mx, 16, 32);   // merge M 0..7 with 8..15
                mx = fmaxf(mx, other);
                if (half == 0) v[(size_t)(state0 + f) * Nn + col] = mx;
            }
    }
}

// ---------------- heads: v[4096x256] -> shoot/scratch/place/pick ----------------
__global__ __launch_bounds__(256) void heads_kernel(const float* __restrict__ v,
                                                    const float* __restrict__ Ws, const float* __restrict__ bs,
                                                    const float* __restrict__ Wc, const float* __restrict__ bc,
                                                    const float* __restrict__ Wp, const float* __restrict__ bp,
                                                    const float* __restrict__ Wk, const float* __restrict__ bk,
                                                    float* __restrict__ out) {
    int t = blockIdx.x * blockDim.x + threadIdx.x;
    if (t >= NUM_STATES * 14) return;
    int state = t / 14, j = t % 14;
    const float* W; const float* b; int col, ncol;
    if (j < 3)      { W = Ws; b = bs; col = j;     ncol = 3; }
    else if (j < 6) { W = Wc; b = bc; col = j - 3; ncol = 3; }
    else if (j < 8) { W = Wp; b = bp; col = j - 6; ncol = 2; }
    else            { W = Wk; b = bk; col = j - 8; ncol = 6; }

    const float* vr = v + (size_t)state * 256;
    float acc = b[col];
    for (int k = 0; k < 256; ++k) acc += vr[k] * W[k * ncol + col];

    float r = acc;
    if (j == 2 || j == 5)  r = fast_sigmoid(acc);
    else if (j == 4)       r = -fast_exp(acc);
    else if (j == 6)       r = 0.5f * fast_sigmoid(acc);
    else if (j == 7)       r = 0.2f * fast_sigmoid(acc) + 0.8f;

    float* dst;
    if (j < 3)      dst = out +                  (size_t)state * 3 + j;
    else if (j < 6) dst = out + NUM_STATES * 3 + (size_t)state * 3 + (j - 3);
    else if (j < 8) dst = out + NUM_STATES * 6 + (size_t)state * 2 + (j - 6);
    else            dst = out + NUM_STATES * 8 + (size_t)state * 6 + (j - 8);
    *dst = r;
}

// ---------------- launch ----------------
extern "C" void kernel_launch(void* const* d_in, const int* in_sizes, int n_in,
                              void* d_out, int out_size, void* d_ws, size_t ws_size,
                              hipStream_t stream) {
    const float* balls = (const float*)d_in[0];
    const float* W1 = (const float*)d_in[1];  const float* b1 = (const float*)d_in[2];
    const float* W2 = (const float*)d_in[3];  const float* b2 = (const float*)d_in[4];
    const float* W3 = (const float*)d_in[5];  const float* b3 = (const float*)d_in[6];
    const float* Ws = (const float*)d_in[7];  const float* bs = (const float*)d_in[8];
    const float* Wc = (const float*)d_in[9];  const float* bc = (const float*)d_in[10];
    const float* Wp = (const float*)d_in[11]; const float* bp = (const float*)d_in[12];
    const float* Wk = (const float*)d_in[13]; const float* bk = (const float*)d_in[14];
    float* out = (float*)d_out;

    char* ws = (char*)d_ws;
    size_t off = 0;
    __hip_bfloat16* Abf = (__hip_bfloat16*)(ws + off); off += (size_t)NROWS * 64 * 2;       // 8 MB
    __hip_bfloat16* Wt1 = (__hip_bfloat16*)(ws + off); off += (size_t)512 * 64 * 2;         // 64 KB
    __hip_bfloat16* Wt2 = (__hip_bfloat16*)(ws + off); off += (size_t)512 * 512 * 2;        // 512 KB
    __hip_bfloat16* Wt3 = (__hip_bfloat16*)(ws + off); off += (size_t)256 * 512 * 2;        // 256 KB
    __hip_bfloat16* h1  = (__hip_bfloat16*)(ws + off); off += (size_t)NROWS * 512 * 2;      // 64 MB
    __hip_bfloat16* h2  = (__hip_bfloat16*)(ws + off); off += (size_t)NROWS * 512 * 2;      // 64 MB
    float*          vbuf= (float*)(ws + off);          off += (size_t)NUM_STATES * 256 * 4; // 4 MB

    // 1) pack to bf16
    pack_balls<<<(NROWS * 64) / 256, 256, 0, stream>>>(balls, Abf);
    pack_wt<<<(512 * 64 + 255) / 256, 256, 0, stream>>>(W1, Wt1, 36, 512, 64);
    pack_wt<<<(512 * 512 + 255) / 256, 256, 0, stream>>>(W2, Wt2, 512, 512, 512);
    pack_wt<<<(256 * 512 + 255) / 256, 256, 0, stream>>>(W3, Wt3, 512, 256, 512);

    // 2) MLP layers: 256 M-rows per block -> 256 blocks
    gemm_tanh_bf16<64> <<<NROWS / 256, 256, 0, stream>>>(Abf, Wt1, b1, h1, NROWS, 64, 512);
    gemm_tanh_bf16<128><<<NROWS / 256, 256, 0, stream>>>(h1, Wt2, b2, h2, NROWS, 512, 512);
    gemm_tanh_pool<128><<<NROWS / 256, 256, 0, stream>>>(h2, Wt3, b3, vbuf, NROWS, 512, 256);

    // 3) heads
    heads_kernel<<<(NUM_STATES * 14 + 255) / 256, 256, 0, stream>>>(
        vbuf, Ws, bs, Wc, bc, Wp, bp, Wk, bk, out);
}
